// GRU_50328426774675
// MI455X (gfx1250) — matrix operand order repaired and checked by reading, not compiled, not run on hardware
//
#include <hip/hip_runtime.h>
#include <hip/hip_bf16.h>
#include <math.h>

typedef __bf16 bf16_t;
typedef __attribute__((ext_vector_type(16))) __bf16 v16bf;
typedef __attribute__((ext_vector_type(8)))  __bf16 v8bf;
typedef __attribute__((ext_vector_type(8)))  float  v8f;

constexpr int I_DIM = 2048;
constexpr int H_DIM = 2048;
constexpr int O_DIM = 2048;
constexpr int T_DIM = 4096;
constexpr int NB_REC = 64;     // persistent blocks for the recurrence

// ---------------------------------------------------------------- conversions
__global__ void cvt_x_kernel(const float* __restrict__ x, bf16_t* __restrict__ xb, int n) {
    int i = blockIdx.x * blockDim.x + threadIdx.x;
    if (i < n) xb[i] = (bf16_t)x[i];
}

// Split W [H, I+H] (fp32, row-major) into bf16 Wx [H,I] and Wh [H,H]
__global__ void cvt_w_kernel(const float* __restrict__ W,
                             bf16_t* __restrict__ wx, bf16_t* __restrict__ wh) {
    int idx = blockIdx.x * blockDim.x + threadIdx.x;
    const int total = H_DIM * (I_DIM + H_DIM);
    if (idx >= total) return;
    int row = idx / (I_DIM + H_DIM);
    int col = idx - row * (I_DIM + H_DIM);
    bf16_t v = (bf16_t)W[idx];
    if (col < I_DIM) wx[(size_t)row * I_DIM + col] = v;
    else             wh[(size_t)row * H_DIM + (col - I_DIM)] = v;
}

// ------------------------------------------------- phase 1: Gx = x @ Wx^T + b
// One wave computes a 16(M=t) x 64(N=j) strip: one A fragment feeds 4 WMMAs.
__global__ void __launch_bounds__(256) gate_x_gemm(
    const bf16_t* __restrict__ xb,                       // [T, I]
    const bf16_t* __restrict__ wxr, const bf16_t* __restrict__ wxz,
    const bf16_t* __restrict__ wxt,                      // [H, I] each
    const float* __restrict__ br, const float* __restrict__ bz,
    const float* __restrict__ bt,
    float* __restrict__ gr, float* __restrict__ gz, float* __restrict__ gt)
{
    const int wave = threadIdx.x >> 5;
    const int lane = threadIdx.x & 31;
    const int hlf  = lane >> 4;        // lane half: 0 or 1
    const int l16  = lane & 15;

    const int t0    = blockIdx.x * 16;                   // T/16 = 256 tiles
    const int jbase = (blockIdx.y * 8 + wave) * 64;      // H/(64*8) = 4 super-strips
    const int gate  = blockIdx.z;

    const bf16_t* wx  = (gate == 0) ? wxr : (gate == 1) ? wxz : wxt;
    const float*  bia = (gate == 0) ? br  : (gate == 1) ? bz  : bt;
    float*        gout= (gate == 0) ? gr  : (gate == 1) ? gz  : gt;

    v8f c[4];
    #pragma unroll
    for (int n = 0; n < 4; ++n) c[n] = (v8f){0.f,0.f,0.f,0.f,0.f,0.f,0.f,0.f};

    const bf16_t* arow = xb + (size_t)(t0 + l16) * I_DIM;      // A: lane row = M
    const bf16_t* brow[4];
    #pragma unroll
    for (int n = 0; n < 4; ++n)
        brow[n] = wx + (size_t)(jbase + 16 * n + l16) * I_DIM; // B: lane col = N

    #pragma unroll 2
    for (int kk = 0; kk < I_DIM; kk += 32) {
        // A 16x32 bf16: element e -> K = e + 8*hlf + (e>=8 ? 8 : 0)
        v8bf alo = *(const v8bf*)(arow + kk + 8 * hlf);        // e=0..7
        v8bf ahi = *(const v8bf*)(arow + kk + 16 + 8 * hlf);   // e=8..15
        v16bf a;
        #pragma unroll
        for (int e = 0; e < 8; ++e) { a[e] = alo[e]; a[e + 8] = ahi[e]; }
        #pragma unroll
        for (int n = 0; n < 4; ++n) {
            // B 32x16 bf16: element e -> K = kk + e + 8*hlf + (e>=8 ? 8 : 0)
            v8bf blo = *(const v8bf*)(brow[n] + kk + 8 * hlf);
            v8bf bhi = *(const v8bf*)(brow[n] + kk + 16 + 8 * hlf);
            v16bf b;
            #pragma unroll
            for (int e = 0; e < 8; ++e) { b[e] = blo[e]; b[e + 8] = bhi[e]; }
            c[n] = __builtin_amdgcn_wmma_f32_16x16x32_bf16(
                       false, a, false, b, (short)0, c[n], false, false);
        }
    }

    #pragma unroll
    for (int n = 0; n < 4; ++n) {
        const int jcol = jbase + 16 * n + l16;
        const float bj = bia[jcol];
        #pragma unroll
        for (int v = 0; v < 8; ++v) {
            int t = t0 + v + 8 * hlf;   // C: element v -> M = v + 8*hlf, N = l16
            // write-once stream: keep it out of L2's hot set
            __builtin_nontemporal_store(c[n][v] + bj,
                                        &gout[(size_t)t * H_DIM + jcol]);
        }
    }
}

// --------------------------------------------------- phase 2: the recurrence
__device__ __forceinline__ void grid_barrier(unsigned int* cnt) {
    __syncthreads();
    if (threadIdx.x == 0) {
        __threadfence();
        unsigned int nb = gridDim.x;
        unsigned int ticket = atomicAdd(cnt, 1u) + 1u;
        unsigned int goal = ((ticket + nb - 1u) / nb) * nb;  // end of this round
        while (__hip_atomic_load(cnt, __ATOMIC_RELAXED,
                                 __HIP_MEMORY_SCOPE_AGENT) < goal) {
            __builtin_amdgcn_s_sleep(1);
        }
        __threadfence();
    }
    __syncthreads();
}

__device__ __forceinline__ float wave_reduce(float acc) {
    #pragma unroll
    for (int off = 16; off > 0; off >>= 1) acc += __shfl_xor(acc, off, 32);
    return acc;
}

__global__ void __launch_bounds__(256) gru_recurrent(
    const bf16_t* __restrict__ whr, const bf16_t* __restrict__ whz,
    const bf16_t* __restrict__ wht,                        // [H,H] bf16
    const float* __restrict__ gr, const float* __restrict__ gz,
    const float* __restrict__ gt,                          // [T,H] fp32 (bias folded)
    float* __restrict__ h, float* __restrict__ rh, float* __restrict__ zg,
    unsigned int* __restrict__ cnt)
{
    __shared__ float hs[H_DIM];
    __shared__ float rhs[H_DIM];
    const int wave = threadIdx.x >> 5;
    const int lane = threadIdx.x & 31;
    const int rowBase = blockIdx.x * 32;   // 64 blocks x 32 rows = 2048

    for (int t = 0; t < T_DIM; ++t) {
        // stage h into LDS
        for (int i = threadIdx.x; i < H_DIM; i += 256) hs[i] = h[i];
        __syncthreads();

        // ---- stage A: r and z rows (4 r-rows + 4 z-rows per wave)
        for (int rr = 0; rr < 8; ++rr) {
            const int gate = rr >> 2;                       // 0=r, 1=z
            const int j = rowBase + wave * 4 + (rr & 3);
            const bf16_t* wrow = ((gate == 0) ? whr : whz) + (size_t)j * H_DIM;
            // prefetch the next row on this wave's critical path (L2-resident)
            if (rr < 7) {
                const int gn = (rr + 1) >> 2;
                const int jn = rowBase + wave * 4 + ((rr + 1) & 3);
                const bf16_t* wnext = ((gn == 0) ? whr : whz) + (size_t)jn * H_DIM;
                __builtin_prefetch(wnext + (size_t)lane * 64, 0, 3);
            }
            float acc = 0.f;
            #pragma unroll
            for (int ck = 0; ck < H_DIM / 256; ++ck) {      // 8 chunks
                const int k = ck * 256 + lane * 8;
                v8bf w = *(const v8bf*)(wrow + k);
                #pragma unroll
                for (int e = 0; e < 8; ++e) acc += (float)w[e] * hs[k + e];
            }
            acc = wave_reduce(acc);
            if (lane == 0) {
                const float* gsrc = (gate == 0) ? gr : gz;
                float pre = acc + __builtin_nontemporal_load(
                                      &gsrc[(size_t)t * H_DIM + j]);
                float s = 1.0f / (1.0f + __expf(-pre));
                if (gate == 0) rh[j] = s * hs[j];
                else           zg[j] = s;
            }
        }
        grid_barrier(cnt);

        // stage r*h into LDS
        for (int i = threadIdx.x; i < H_DIM; i += 256) rhs[i] = rh[i];
        __syncthreads();

        // ---- stage B: h_tilde and h update (4 rows per wave)
        for (int rr = 0; rr < 4; ++rr) {
            const int j = rowBase + wave * 4 + rr;
            const bf16_t* wrow = wht + (size_t)j * H_DIM;
            if (rr < 3)
                __builtin_prefetch(wrow + H_DIM + (size_t)lane * 64, 0, 3);
            float acc = 0.f;
            #pragma unroll
            for (int ck = 0; ck < H_DIM / 256; ++ck) {
                const int k = ck * 256 + lane * 8;
                v8bf w = *(const v8bf*)(wrow + k);
                #pragma unroll
                for (int e = 0; e < 8; ++e) acc += (float)w[e] * rhs[k + e];
            }
            acc = wave_reduce(acc);
            if (lane == 0) {
                float ht = tanhf(acc + __builtin_nontemporal_load(
                                           &gt[(size_t)t * H_DIM + j]));
                float zz = zg[j];
                h[j] = zz * hs[j] + (1.0f - zz) * ht;       // in-place: own j only
            }
        }
        grid_barrier(cnt);
    }
}

// ------------------------------------------------- final: out = Wo @ h + bo
__global__ void __launch_bounds__(256) final_out(
    const float* __restrict__ Wo, const float* __restrict__ bo,
    const float* __restrict__ h, float* __restrict__ out)
{
    const int wave = threadIdx.x >> 5;
    const int lane = threadIdx.x & 31;
    const int row = blockIdx.x * 8 + wave;                 // 256 blocks x 8 waves
    const float* wrow = Wo + (size_t)row * H_DIM;
    float acc = 0.f;
    for (int i = 0; i < H_DIM / 32; ++i) {
        const int k = i * 32 + lane;
        acc += wrow[k] * h[k];
    }
    acc = wave_reduce(acc);
    if (lane == 0) {
        out[row]         = acc + bo[row];
        out[O_DIM + row] = h[row];                          // second tuple element
    }
}

// ---------------------------------------------------------------- launcher
extern "C" void kernel_launch(void* const* d_in, const int* in_sizes, int n_in,
                              void* d_out, int out_size, void* d_ws, size_t ws_size,
                              hipStream_t stream) {
    const float* x  = (const float*)d_in[0];   // [T, I]
    const float* h0 = (const float*)d_in[1];   // [H]
    const float* Wr = (const float*)d_in[2];
    const float* br = (const float*)d_in[3];
    const float* Wz = (const float*)d_in[4];
    const float* bz = (const float*)d_in[5];
    const float* Wt = (const float*)d_in[6];
    const float* bt = (const float*)d_in[7];
    const float* Wo = (const float*)d_in[8];
    const float* bo = (const float*)d_in[9];
    float* out = (float*)d_out;

    char* ws = (char*)d_ws;
    size_t off = 0;
    auto alloc = [&](size_t bytes) -> void* {
        void* p = ws + off;
        off = (off + bytes + 255) & ~(size_t)255;
        return p;
    };

    bf16_t* xb  = (bf16_t*)alloc((size_t)T_DIM * I_DIM * sizeof(bf16_t));
    bf16_t* wxr = (bf16_t*)alloc((size_t)H_DIM * I_DIM * sizeof(bf16_t));
    bf16_t* wxz = (bf16_t*)alloc((size_t)H_DIM * I_DIM * sizeof(bf16_t));
    bf16_t* wxt = (bf16_t*)alloc((size_t)H_DIM * I_DIM * sizeof(bf16_t));
    bf16_t* whr = (bf16_t*)alloc((size_t)H_DIM * H_DIM * sizeof(bf16_t));
    bf16_t* whz = (bf16_t*)alloc((size_t)H_DIM * H_DIM * sizeof(bf16_t));
    bf16_t* wht = (bf16_t*)alloc((size_t)H_DIM * H_DIM * sizeof(bf16_t));
    float*  gr  = (float*)alloc((size_t)T_DIM * H_DIM * sizeof(float));
    float*  gz  = (float*)alloc((size_t)T_DIM * H_DIM * sizeof(float));
    float*  gt  = (float*)alloc((size_t)T_DIM * H_DIM * sizeof(float));
    float*  h   = (float*)alloc(H_DIM * sizeof(float));
    float*  rh  = (float*)alloc(H_DIM * sizeof(float));
    float*  zg  = (float*)alloc(H_DIM * sizeof(float));
    unsigned int* cnt = (unsigned int*)alloc(256);

    // deterministic state for every (re)play
    hipMemsetAsync(cnt, 0, 256, stream);
    hipMemcpyAsync(h, h0, H_DIM * sizeof(float), hipMemcpyDeviceToDevice, stream);

    // conversions to bf16
    {
        const int n = T_DIM * I_DIM;
        cvt_x_kernel<<<(n + 255) / 256, 256, 0, stream>>>(x, xb, n);
        const int m = H_DIM * (I_DIM + H_DIM);
        cvt_w_kernel<<<(m + 255) / 256, 256, 0, stream>>>(Wr, wxr, whr);
        cvt_w_kernel<<<(m + 255) / 256, 256, 0, stream>>>(Wz, wxz, whz);
        cvt_w_kernel<<<(m + 255) / 256, 256, 0, stream>>>(Wt, wxt, wht);
    }

    // phase 1: parallel WMMA GEMMs for the x-contribution of all gates
    dim3 g1(T_DIM / 16, (H_DIM / 16) / 8 / 4, 3);   // (256, 4, 3): 64-wide strips
    gate_x_gemm<<<g1, 256, 0, stream>>>(xb, wxr, wxz, wxt, br, bz, bt, gr, gz, gt);

    // phase 2: persistent sequential recurrence with grid barriers
    gru_recurrent<<<NB_REC, 256, 0, stream>>>(whr, whz, wht, gr, gz, gt,
                                              h, rh, zg, cnt);

    // final projection + hidden copy-out
    final_out<<<O_DIM / 8, 256, 0, stream>>>(Wo, bo, h, out);
}
